// ArcMarginProduct_70866960384684
// MI455X (gfx1250) — matrix operand order repaired and checked by reading, not compiled, not run on hardware
//
#include <hip/hip_runtime.h>
#include <stdint.h>

// ---------------------------------------------------------------------------
// ArcMarginProduct forward for MI455X (gfx1250, wave32, WMMA).
//   cosine = normalize(x) @ normalize(W)^T   (1024x512 @ 512x100000)
//   margin epilogue + *S, label-indexed select.
// Strategy: pre-normalize both operands into bf16 (ws), then a bf16 WMMA GEMM
// with fp32 accumulation. bf16 weights (102.4MB) are L2-resident (192MB L2),
// so HBM traffic ~= 205MB W read + 102MB W-bf16 write + 410MB output write,
// ~31us at 23.3TB/s; bf16 matrix compute (~105 GFLOP) lands near the same,
// i.e. balanced roofline. fp32 WMMA (16x16x4) would be ~8x slower on compute.
// Workspace requirement: (M*D + C*D) * 2 bytes ~= 103.5 MB.
// ---------------------------------------------------------------------------

typedef __attribute__((ext_vector_type(16))) __bf16 v16bf;
typedef __attribute__((ext_vector_type(8)))  __bf16 v8bf;
typedef __attribute__((ext_vector_type(8)))  float  v8f;
typedef __attribute__((ext_vector_type(4)))  float  v4f;
typedef __attribute__((ext_vector_type(4)))  unsigned short v4u16;

#define ARC_S            30.0f
#define ARC_COS_M        0.8775825618903728f   // cos(0.5)
#define ARC_SIN_M        0.479425538604203f    // sin(0.5)
#define ARC_TH          (-0.8775825618903728f) // cos(pi - 0.5)
#define ARC_FALLBACK_SUB 7.191383079063045f    // S * sin(m) * m

__device__ __forceinline__ unsigned short f32_to_bf16_rne(float f) {
  unsigned int u = __float_as_uint(f);
  u += 0x7FFFu + ((u >> 16) & 1u);   // round-to-nearest-even
  return (unsigned short)(u >> 16);
}

// One wave32 per row: L2-normalize a [rows x D] fp32 matrix and emit bf16.
// Requires D % 128 == 0 (D = 512 here).
__global__ void rownorm_to_bf16(const float* __restrict__ src,
                                unsigned short* __restrict__ dst,
                                int rows, int D) {
  const int gtid = blockIdx.x * blockDim.x + threadIdx.x;
  const int row  = gtid >> 5;
  const int lane = threadIdx.x & 31;
  if (row >= rows) return;

  const float* r = src + (size_t)row * D;
  float s = 0.0f;
  for (int k = lane * 4; k < D; k += 128) {        // coalesced float4 loads
    v4f v = *(const v4f*)(r + k);
    s += v.x * v.x + v.y * v.y + v.z * v.z + v.w * v.w;
  }
  // wave32 butterfly reduction
  #pragma unroll
  for (int off = 16; off > 0; off >>= 1)
    s += __shfl_xor(s, off, 32);

  const float inv = 1.0f / fmaxf(sqrtf(s), 1e-12f);

  unsigned short* o = dst + (size_t)row * D;
  for (int k = lane * 4; k < D; k += 128) {
    v4f v = *(const v4f*)(r + k);
    v4u16 b;
    b.x = f32_to_bf16_rne(v.x * inv);
    b.y = f32_to_bf16_rne(v.y * inv);
    b.z = f32_to_bf16_rne(v.z * inv);
    b.w = f32_to_bf16_rne(v.w * inv);
    *(v4u16*)(o + k) = b;
  }
}

// ---------------------------------------------------------------------------
// WMMA GEMM + ArcFace epilogue.
// Block: 256 threads = 8 waves arranged 2 (M) x 4 (N).
// Block tile 128x128; wave tile 64x32 = 4 A-frags x 2 B-frags.
// Per K-step (32): 8x v_wmma_f32_16x16x32_bf16.
// Requires M % 128 == 0, D % 32 == 0. N (=C) is bounds-handled.
// ---------------------------------------------------------------------------
__global__ __launch_bounds__(256) void arcface_gemm_wmma(
    const __bf16* __restrict__ A,    // [M, D] row-major, normalized
    const __bf16* __restrict__ W,    // [C, D] row-major, normalized (acts as B^T)
    const int*    __restrict__ label,// [M]
    float*        __restrict__ out,  // [M, C]
    int M, int C, int D) {

  const int lane = threadIdx.x & 31;
  const int wv   = threadIdx.x >> 5;   // 0..7
  const int rr   = lane & 15;          // row-in-frag (A) / col-in-frag (B,C)
  const int h    = lane >> 4;          // half-wave select

  const int waveM = wv & 1;            // 2 wave rows
  const int waveN = wv >> 1;           // 4 wave cols
  const int gm0 = blockIdx.y * 128 + waveM * 64;   // 4 m-frags of 16
  const int gc0 = blockIdx.x * 128 + waveN * 32;   // 2 n-frags of 16

  // A fragment base pointers (ISA 16-bit A 16x32 layout):
  //   lane holds row (rr), chunks of 8 elems at k0 + 8h and k0 + 16 + 8h.
  const __bf16* aptr[4];
  #pragma unroll
  for (int mi = 0; mi < 4; ++mi)
    aptr[mi] = A + (size_t)(gm0 + mi * 16 + rr) * D + h * 8;

  // B fragment base pointers (dense 16-bit B 32x16 layout):
  //   lane holds column (rr), chunks of 8 elems at k0 + 8h and k0 + 16 + 8h.
  //   Clamp out-of-range columns; their results are discarded at store time.
  const __bf16* bptr[2];
  #pragma unroll
  for (int ni = 0; ni < 2; ++ni) {
    int col = gc0 + ni * 16 + rr;
    if (col > C - 1) col = C - 1;
    bptr[ni] = W + (size_t)col * D + h * 8;
  }

  v8f acc[4][2] = {};   // fp32 accumulators, 16x16 each

  #pragma unroll 2
  for (int k0 = 0; k0 < D; k0 += 32) {
    v16bf af[4], bfr[2];
    #pragma unroll
    for (int mi = 0; mi < 4; ++mi) {
      v8bf lo = *(const v8bf*)(aptr[mi] + k0);
      v8bf hi = *(const v8bf*)(aptr[mi] + k0 + 16);
      af[mi] = __builtin_shufflevector(lo, hi, 0, 1, 2, 3, 4, 5, 6, 7,
                                               8, 9, 10, 11, 12, 13, 14, 15);
    }
    #pragma unroll
    for (int ni = 0; ni < 2; ++ni) {
      v8bf lo = *(const v8bf*)(bptr[ni] + k0);
      v8bf hi = *(const v8bf*)(bptr[ni] + k0 + 16);
      bfr[ni] = __builtin_shufflevector(lo, hi, 0, 1, 2, 3, 4, 5, 6, 7,
                                                8, 9, 10, 11, 12, 13, 14, 15);
    }
    #pragma unroll
    for (int mi = 0; mi < 4; ++mi)
      #pragma unroll
      for (int ni = 0; ni < 2; ++ni)
        acc[mi][ni] = __builtin_amdgcn_wmma_f32_16x16x32_bf16(
            /*neg_a=*/false, af[mi], /*neg_b=*/false, bfr[ni],
            /*c_mod=*/(short)0, acc[mi][ni],
            /*reuse_a=*/false, /*reuse_b=*/false);
  }

  // Epilogue: ArcFace margin. C/D layout: element v <-> row (v + 8h), col rr.
  #pragma unroll
  for (int mi = 0; mi < 4; ++mi) {
    #pragma unroll
    for (int ni = 0; ni < 2; ++ni) {
      const int gc = gc0 + ni * 16 + rr;
      if (gc < C) {
        #pragma unroll
        for (int v = 0; v < 8; ++v) {
          const int gm = gm0 + mi * 16 + v + h * 8;
          const float c  = acc[mi][ni][v];
          float s2 = 1.0f - c * c;
          s2 = fminf(fmaxf(s2, 0.0f), 1.0f);
          const float sn  = sqrtf(s2);
          const float phi = c * ARC_COS_M - sn * ARC_SIN_M;
          const float sel = (c > ARC_TH) ? phi : (c - ARC_FALLBACK_SUB);
          const float o   = ((label[gm] == gc) ? sel : c) * ARC_S;
          out[(size_t)gm * C + gc] = o;
        }
      }
    }
  }
}

extern "C" void kernel_launch(void* const* d_in, const int* in_sizes, int n_in,
                              void* d_out, int out_size, void* d_ws, size_t ws_size,
                              hipStream_t stream) {
  const float* inp    = (const float*)d_in[0];   // [B, D] fp32
  const int*   label  = (const int*)d_in[1];     // [B] int32
  const float* weight = (const float*)d_in[2];   // [C, D] fp32
  float*       out    = (float*)d_out;           // [B, C] fp32

  const int B = in_sizes[1];
  const int D = in_sizes[0] / B;                 // 512
  const int C = in_sizes[2] / D;                 // 100000

  // Workspace: normalized bf16 A then normalized bf16 W.
  unsigned short* wsA = (unsigned short*)d_ws;                 // B*D bf16
  unsigned short* wsW = wsA + (size_t)B * D;                   // C*D bf16

  // 1) normalize input -> bf16 (one wave per row, 8 rows per 256-thread block)
  rownorm_to_bf16<<<(B + 7) / 8, 256, 0, stream>>>(inp, wsA, B, D);
  // 2) normalize weight -> bf16
  rownorm_to_bf16<<<(C + 7) / 8, 256, 0, stream>>>(weight, wsW, C, D);

  // 3) WMMA GEMM + margin epilogue
  dim3 grid((C + 127) / 128, B / 128);
  arcface_gemm_wmma<<<grid, 256, 0, stream>>>(
      (const __bf16*)wsA, (const __bf16*)wsW, label, out, B, C, D);
}
